// Fp8Linear_33432025432368
// MI455X (gfx1250) — 8-bit matrix operand order repaired and checked by reading, compiled, not run on hardware
//
#include <hip/hip_runtime.h>

typedef __attribute__((ext_vector_type(16))) int   v16i;
typedef __attribute__((ext_vector_type(8)))  float v8f;


__device__ inline float clamp_e4m3(float v) {
    v = fminf(v, 448.0f);
    v = fmaxf(v, -448.0f);
    return v;
}

#if !__has_builtin(__builtin_amdgcn_cvt_pk_fp8_f32)
__device__ inline unsigned int f32_to_e4m3_byte(float x) {
    unsigned s = __float_as_uint(x) >> 31;
    float ax = __builtin_fabsf(x);
    if (ax != ax) return (s << 7) | 0x7F;
    if (ax > 448.0f) return (s << 7) | 0x7E;
    if (ax < 0.0009765625f) return (s << 7);
    int e; float m = __builtin_frexpf(ax, &e);
    int te = e - 1;
    if (te < -6) {
        unsigned mi = (unsigned)__builtin_rintf(ax * 512.0f);
        return (s << 7) | (mi & 7);
    }
    unsigned mi = (unsigned)__builtin_rintf(m * 16.0f);
    if (mi == 16) { mi = 8; te += 1; }
    if (te > 8) return (s << 7) | 0x7E;
    return (s << 7) | (((unsigned)(te + 7)) << 4) | (mi & 7);
}
#endif

__global__ __launch_bounds__(256)
void fp8lin_quant_kernel(const float* __restrict__ x,
                         const float* __restrict__ s_in,
                         unsigned char* __restrict__ aq,
                         long long n8) {
    long long i = (long long)blockIdx.x * blockDim.x + threadIdx.x;
    if (i >= n8) return;
    float inv = 1.0f / s_in[0];
    const float4* p = (const float4*)x + i * 2;
    float4 v0 = p[0], v1 = p[1];
    float a0 = clamp_e4m3(v0.x * inv), a1 = clamp_e4m3(v0.y * inv);
    float a2 = clamp_e4m3(v0.z * inv), a3 = clamp_e4m3(v0.w * inv);
    float a4 = clamp_e4m3(v1.x * inv), a5 = clamp_e4m3(v1.y * inv);
    float a6 = clamp_e4m3(v1.z * inv), a7 = clamp_e4m3(v1.w * inv);
    unsigned int lo, hi;
#if __has_builtin(__builtin_amdgcn_cvt_pk_fp8_f32)
    int w0 = __builtin_amdgcn_cvt_pk_fp8_f32(a0, a1, 0, false);
    w0     = __builtin_amdgcn_cvt_pk_fp8_f32(a2, a3, w0, true);
    int w1 = __builtin_amdgcn_cvt_pk_fp8_f32(a4, a5, 0, false);
    w1     = __builtin_amdgcn_cvt_pk_fp8_f32(a6, a7, w1, true);
    lo = (unsigned int)w0; hi = (unsigned int)w1;
#else
    lo = f32_to_e4m3_byte(a0)         | (f32_to_e4m3_byte(a1) << 8) |
         (f32_to_e4m3_byte(a2) << 16) | (f32_to_e4m3_byte(a3) << 24);
    hi = f32_to_e4m3_byte(a4)         | (f32_to_e4m3_byte(a5) << 8) |
         (f32_to_e4m3_byte(a6) << 16) | (f32_to_e4m3_byte(a7) << 24);
#endif
    ((uint2*)aq)[i] = make_uint2(lo, hi);
}


#define KSTEP    128
#define LDS_ROW  144
#define LDS_TILE (128 * LDS_ROW)

__device__ inline void async_b128_to_lds(unsigned lds_off, const void* gaddr) {
    asm volatile("global_load_async_to_lds_b128 %0, %1, off"
                 :: "v"(lds_off), "v"((unsigned long long)(size_t)gaddr)
                 : "memory");
}

__device__ inline void wait_async0() {
#if __has_builtin(__builtin_amdgcn_s_wait_asynccnt)
    __builtin_amdgcn_s_wait_asynccnt(0);
#else
    asm volatile("s_wait_asynccnt 0" ::: "memory");
#endif
}

__device__ inline void issue_btile(const unsigned char* __restrict__ Wq,
                                   long long K, int n0, int k,
                                   unsigned lds_buf_off, int tid) {
#pragma unroll
    for (int j = 0; j < 4; ++j) {
        int chunk = tid + 256 * j;
        int row   = chunk >> 3;
        int kc    = (chunk & 7) * 16;
        const unsigned char* g = Wq + (long long)(n0 + row) * K + k + kc;
        async_b128_to_lds(lds_buf_off + row * LDS_ROW + kc, g);
    }
}

__device__ inline v16i load_a_frag(const unsigned char* __restrict__ ap) {
    v16i a;
#pragma unroll
    for (int j = 0; j < 8; ++j) {
        uint2 d = *(const uint2*)(ap + (j >> 1) * 32 + (j & 1) * 8);
        a[2 * j]     = (int)d.x;
        a[2 * j + 1] = (int)d.y;
    }
    return a;
}

__global__ __launch_bounds__(256)
void fp8lin_gemm_kernel(const unsigned char* __restrict__ Aq,
                        const unsigned char* __restrict__ Wq,
                        const float* __restrict__ bias,
                        const float* __restrict__ s_in,
                        const float* __restrict__ s_w,
                        float* __restrict__ out,
                        int M, int N, int K) {
    __shared__ __align__(16) unsigned char smem[2 * LDS_TILE];

    const int tid   = threadIdx.x;
    const int wave  = tid >> 5;
    const int lane  = tid & 31;
    const int lrow  = lane & 15;
    const int lhalf = lane >> 4;

    const int m0 = blockIdx.y * 256 + wave * 32;
    const int n0 = blockIdx.x * 128;

    const unsigned smem_lo = (unsigned)(size_t)(void*)smem;

    v8f acc0[8] = {};
    v8f acc1[8] = {};

    const unsigned char* arow0 = Aq + (long long)(m0 + lrow) * K + lhalf * 16;
    const unsigned char* arow1 = arow0 + 16LL * K;
    const int bkoff = lhalf * 16;
    const int nk = K / KSTEP;

    issue_btile(Wq, K, n0, 0, smem_lo, tid);
    v16i a0 = load_a_frag(arow0);
    v16i a1 = load_a_frag(arow1);

    for (int kt = 0; kt < nk; ++kt) {
        wait_async0();
        __syncthreads();

        if (kt + 1 < nk)
            issue_btile(Wq, K, n0, (kt + 1) * KSTEP,
                        smem_lo + ((unsigned)((kt + 1) & 1)) * LDS_TILE, tid);

        const int kpre = (kt + 1 < nk ? kt + 1 : kt) * KSTEP;
        v16i an0 = load_a_frag(arow0 + kpre);
        v16i an1 = load_a_frag(arow1 + kpre);

        const unsigned char* sbase = smem + (kt & 1) * LDS_TILE;
#pragma unroll
        for (int nt = 0; nt < 8; ++nt) {
            const unsigned char* sb =
                sbase + (nt * 16 + lrow) * LDS_ROW + bkoff;
            v16i b;
#pragma unroll
            for (int j = 0; j < 4; ++j) {
                int4 d = *(const int4*)(sb + j * 32);
                b[4 * j]     = d.x;
                b[4 * j + 1] = d.y;
                b[4 * j + 2] = d.z;
                b[4 * j + 3] = d.w;
            }
            acc0[nt] = __builtin_amdgcn_wmma_f32_16x16x128_fp8_fp8(
                a0, b, (short)0, acc0[nt], false, false);
            acc1[nt] = __builtin_amdgcn_wmma_f32_16x16x128_fp8_fp8(
                a1, b, (short)0, acc1[nt], false, false);
        }
        a0 = an0;
        a1 = an1;
    }

    const float scale = s_in[0] * s_w[0];
#pragma unroll
    for (int nt = 0; nt < 8; ++nt) {
        const int n = n0 + nt * 16 + lrow;
        const float bz = bias[n];
#pragma unroll
        for (int r = 0; r < 8; ++r) {
            long long mA = (long long)(m0 + r + lhalf * 8);
            long long mB = mA + 16;
            __builtin_nontemporal_store(acc0[nt][r] * scale + bz,
                                        &out[mA * N + n]);
            __builtin_nontemporal_store(acc1[nt][r] * scale + bz,
                                        &out[mB * N + n]);
        }
    }
}


extern "C" void kernel_launch(void* const* d_in, const int* in_sizes, int n_in,
                              void* d_out, int out_size, void* d_ws, size_t ws_size,
                              hipStream_t stream) {
    const float*         x    = (const float*)d_in[0];
    const unsigned char* wq   = (const unsigned char*)d_in[1];
    const float*         bias = (const float*)d_in[2];
    const float*         s_in = (const float*)d_in[3];
    const float*         s_w  = (const float*)d_in[4];
    float*               out  = (float*)d_out;

    const int       n = in_sizes[2];
    const long long k = (long long)in_sizes[1] / n;
    const long long m = (long long)in_sizes[0] / k;

    unsigned char* aq = (unsigned char*)d_ws;

    const long long n8 = (m * k) / 8;
    fp8lin_quant_kernel<<<(unsigned)((n8 + 255) / 256), 256, 0, stream>>>(
        x, s_in, aq, n8);

    dim3 grid((unsigned)(n / 128), (unsigned)(m / 256));
    fp8lin_gemm_kernel<<<grid, 256, 0, stream>>>(
        aq, wq, bias, s_in, s_w, out, (int)m, n, (int)k);
}
